// CFC_V3_Model_46016279610141
// MI455X (gfx1250) — matrix operand order repaired and checked by reading, not compiled, not run on hardware
//
#include <hip/hip_runtime.h>
#include <hip/hip_bf16.h>

// ---------------- problem constants ----------------
#define B_    128
#define T_    512
#define S_    30
#define F_    64
#define KS_   5
#define H_    128
#define NC_   18
#define L0_   254
#define L1_   125
#define L2_   61
#define L3_   29
#define NSL_  (B_ * L3_)      // 3712 attention slices
#define FS_   (F_ * S_)       // 1920

typedef __attribute__((ext_vector_type(16))) _Float16 v16h;
typedef __attribute__((ext_vector_type(8)))  _Float16 v8h;
typedef __attribute__((ext_vector_type(8)))  float    v8f;

// ---------------- WMMA fragment helpers (wave32, 16x16x32 f16) ------------
// Per-lane K runs are contiguous:
//   A (16x32): elems 0-7 = K hi*8..hi*8+7 ; elems 8-15 = K 16+hi*8..+7
//   B (32x16): same as A: elems 0-7 = K hi*8..hi*8+7 ; elems 8-15 = K 16+hi*8..+7
__device__ __forceinline__ int a_row(int lane) { return lane & 15; }
__device__ __forceinline__ int b_col(int lane) { return lane & 15; }
__device__ __forceinline__ int cd_col(int lane) { return lane & 15; }
__device__ __forceinline__ int cd_row(int lane, int r) { return r + ((lane >> 4) ? 8 : 0); }

// p points at element K=0 of this lane's row/col; row must be contiguous in K.
__device__ __forceinline__ v16h load_a_row(const _Float16* p, int lane) {
  const int hb = ((lane >> 4) & 1) * 8;
  v8h lo = *(const v8h*)(p + hb);
  v8h hi = *(const v8h*)(p + 16 + hb);
  return __builtin_shufflevector(lo, hi, 0, 1, 2, 3, 4, 5, 6, 7,
                                 8, 9, 10, 11, 12, 13, 14, 15);
}
__device__ __forceinline__ v16h load_b_row(const _Float16* p, int lane) {
  const int hb = ((lane >> 4) & 1) * 8;
  v8h lo = *(const v8h*)(p + hb);
  v8h hi = *(const v8h*)(p + 16 + hb);
  return __builtin_shufflevector(lo, hi, 0, 1, 2, 3, 4, 5, 6, 7,
                                 8, 9, 10, 11, 12, 13, 14, 15);
}

__device__ __forceinline__ v8f wmma16(v16h a, v16h b, v8f c) {
  return __builtin_amdgcn_wmma_f32_16x16x32_f16(false, a, false, b, (short)0, c,
                                                false, false);
}

// ---------------- utility kernels ----------------
__global__ void k_cvt_f16(const float* __restrict__ s, _Float16* __restrict__ d, int n) {
  int i = blockIdx.x * blockDim.x + threadIdx.x;
  if (i < n) d[i] = (_Float16)s[i];
}

__global__ void k_zero_f16(_Float16* __restrict__ d, int n) {
  int i = blockIdx.x * blockDim.x + threadIdx.x;
  if (i < n) d[i] = (_Float16)0.f;
}

// convs_w layer (F,C,5) f32 -> f16 [f][tap][c] so K (=c) is contiguous
__global__ void k_convw(const float* __restrict__ src, _Float16* __restrict__ dst) {
  int i = blockIdx.x * blockDim.x + threadIdx.x;     // over 64*5*64 dst elems
  if (i >= F_ * KS_ * F_) return;
  int f = i / (KS_ * F_);
  int rem = i - f * (KS_ * F_);
  int tap = rem / F_;
  int c = rem - tap * F_;
  dst[i] = (_Float16)src[(f * F_ + c) * KS_ + tap];
}

// ---------------- conv0 (in-channels = 1) + ReLU + BN -> act f16 [col][64] --
__global__ void k_conv0(const float* __restrict__ x, const float* __restrict__ w0,
                        const float* __restrict__ bias, const float* __restrict__ g,
                        const float* __restrict__ be, const float* __restrict__ mean,
                        const float* __restrict__ var, _Float16* __restrict__ out) {
  int idx = blockIdx.x * blockDim.x + threadIdx.x;   // over B*L0*S columns
  if (idx >= B_ * L0_ * S_) return;
  int bb = idx / (L0_ * S_);
  int rem = idx - bb * (L0_ * S_);
  int t = rem / S_;
  int s = rem - t * S_;
  float xv[KS_];
#pragma unroll
  for (int k = 0; k < KS_; ++k) xv[k] = x[((long)bb * T_ + 2 * t + k) * S_ + s];
  long obase = (long)idx * F_;
#pragma unroll
  for (int fc = 0; fc < F_ / 8; ++fc) {     // vectorized 8-wide stores
    v8h chunk;
#pragma unroll
    for (int j = 0; j < 8; ++j) {
      int f = fc * 8 + j;
      float acc = bias[f];
#pragma unroll
      for (int k = 0; k < KS_; ++k) acc += w0[f * KS_ + k] * xv[k];
      acc = fmaxf(acc, 0.f);
      float inv = g[f] * rsqrtf(var[f] + 1e-5f);
      acc = acc * inv + (be[f] - mean[f] * inv);
      chunk[j] = (_Float16)acc;
    }
    *(v8h*)(out + obase + fc * 8) = chunk;
  }
}

// ------- conv layers 1..3 as implicit GEMM: M=64(out ch), K=320, N=B*Lout*S --
// act layout: [col][64] channels-last, col = (b*Lt + t)*S + s
// Lin/Lout are template params so column decomposition is magic-mul, not div.
template <int Lin, int Lout>
__global__ void k_conv_wmma(const _Float16* __restrict__ in,  // [B*Lin*S][64]
                            const _Float16* __restrict__ w16, // [f][tap][c]
                            const float* __restrict__ bias, const float* __restrict__ g,
                            const float* __restrict__ be, const float* __restrict__ mean,
                            const float* __restrict__ var,
                            _Float16* __restrict__ out) {     // [B*Lout*S][64]
  const int lane = threadIdx.x;
  const int ntile = blockIdx.x;
  const int mtile = blockIdx.y;                       // 0..3 (64 channels)
  const int col = ntile * 16 + b_col(lane);           // output column for B frag
  const int bb = col / (Lout * S_);
  int rem = col - bb * (Lout * S_);
  const int t = rem / S_;
  const int s = rem - t * S_;
  const int fA = mtile * 16 + a_row(lane);
  const _Float16* wrow0 = w16 + (long)fA * KS_ * F_;  // [tap][c]

  v8f acc = {};
#pragma unroll
  for (int tap = 0; tap < KS_; ++tap) {
    const _Float16* irow = in + (((long)bb * Lin + 2 * t + tap) * S_ + s) * F_;
    const _Float16* wrow = wrow0 + tap * F_;
    __builtin_prefetch(irow + 2 * S_ * F_, 0, 0);     // input row two taps ahead
#pragma unroll
    for (int cb = 0; cb < 2; ++cb) {
      v16h a = load_a_row(wrow + cb * 32, lane);
      v16h bf = load_b_row(irow + cb * 32, lane);
      acc = wmma16(a, bf, acc);
    }
  }
  const int colD = ntile * 16 + cd_col(lane);
  const long obase = (long)colD * F_;
#pragma unroll
  for (int r = 0; r < 8; ++r) {
    int f = mtile * 16 + cd_row(lane, r);
    float y = fmaxf(acc[r] + bias[f], 0.f);
    float inv = g[f] * rsqrtf(var[f] + 1e-5f);
    y = y * inv + (be[f] - mean[f] * inv);
    out[obase + f] = (_Float16)y;
  }
}

// ---------------- fused self-attention per (b,l) slice ----------------
// act: [NSL*S][64] channels-last ; att out: [n][f*30+s] (FC1 row-major K)
// LDS layouts chosen so every WMMA operand is contiguous in K:
//   s_xt[s][f]  (x transposed = natural channels-last rows)
//   s_qt[s][o], s_kt[s][o]  (q,k transposed)
//   s_v [f][i]              (v normal)
//   s_btT[j][i]             (beta transposed)
__global__ void __launch_bounds__(256)
k_attn(const _Float16* __restrict__ act, const _Float16* __restrict__ wq16,
       const _Float16* __restrict__ wk16, const _Float16* __restrict__ wv16,
       const float* __restrict__ gamma, _Float16* __restrict__ att) {
  __shared__ alignas(32) _Float16 s_xt[32 * F_];
  __shared__ alignas(32) _Float16 s_qt[32 * F_];
  __shared__ alignas(32) _Float16 s_kt[32 * F_];
  __shared__ alignas(32) _Float16 s_v [F_ * 32];
  __shared__ alignas(32) float    s_sc[32 * 32];
  __shared__ alignas(32) _Float16 s_btT[32 * 32];
  const int n = blockIdx.x;
  const int tid = threadIdx.x;
  const int lane = tid & 31;
  const int wave = tid >> 5;

  // stage x: one v8h per thread (32 rows x 8 chunks)
  {
    int s = tid >> 3, c = (tid & 7) * 8;
    v8h val = {};
    if (s < S_) val = *(const v8h*)(act + ((long)n * S_ + s) * F_ + c);
    *(v8h*)(s_xt + s * F_ + c) = val;
  }
  __syncthreads();

  // phase 1: q,k,v = W @ x   (M=64 -> 4 tiles, N=32 -> 2 tiles, K=64)
  {
    const int mt = wave >> 1, nt = wave & 1;
    const _Float16* W[3] = {wq16, wk16, wv16};
    _Float16* Dt[2] = {s_qt, s_kt};
    const int rowA = mt * 16 + a_row(lane);
    const int colB = nt * 16 + b_col(lane);
#pragma unroll
    for (int mm = 0; mm < 3; ++mm) {
      v8f acc = {};
#pragma unroll
      for (int ksb = 0; ksb < 2; ++ksb) {
        v16h a = load_a_row(W[mm] + rowA * F_ + ksb * 32, lane);
        v16h bf = load_b_row(s_xt + colB * F_ + ksb * 32, lane);
        acc = wmma16(a, bf, acc);
      }
#pragma unroll
      for (int r = 0; r < 8; ++r) {
        int o = mt * 16 + cd_row(lane, r);
        int s = nt * 16 + cd_col(lane);
        if (mm == 2) s_v[o * 32 + s] = (_Float16)acc[r];
        else         Dt[mm][s * F_ + o] = (_Float16)acc[r];
      }
    }
  }
  __syncthreads();

  // phase 2: scores[i][j] = sum_o q[o][i]*k[o][j]  (waves 0..3; reads qT,kT)
  if (wave < 4) {
    const int it = wave >> 1, jt = wave & 1;
    const int i = it * 16 + a_row(lane);
    const int j = jt * 16 + b_col(lane);
    v8f acc = {};
#pragma unroll
    for (int ksb = 0; ksb < 2; ++ksb) {
      v16h a = load_a_row(s_qt + i * F_ + ksb * 32, lane);
      v16h bf = load_b_row(s_kt + j * F_ + ksb * 32, lane);
      acc = wmma16(a, bf, acc);
    }
#pragma unroll
    for (int r = 0; r < 8; ++r)
      s_sc[(it * 16 + cd_row(lane, r)) * 32 + jt * 16 + cd_col(lane)] = acc[r];
  }
  __syncthreads();

  // phase 3: softmax over i (axis=1) per column j; write beta transposed
  if (tid < 32) {
    int j = tid;
    if (j < S_) {
      float mx = -1e30f;
#pragma unroll
      for (int i = 0; i < S_; ++i) mx = fmaxf(mx, s_sc[i * 32 + j]);
      float e[S_], sum = 0.f;
#pragma unroll
      for (int i = 0; i < S_; ++i) { e[i] = __expf(s_sc[i * 32 + j] - mx); sum += e[i]; }
      float inv = 1.f / sum;
#pragma unroll
      for (int i = 0; i < S_; ++i) s_btT[j * 32 + i] = (_Float16)(e[i] * inv);
      s_btT[j * 32 + 30] = (_Float16)0.f;
      s_btT[j * 32 + 31] = (_Float16)0.f;
    } else {
      for (int i = 0; i < 32; ++i) s_btT[j * 32 + i] = (_Float16)0.f;
    }
  }
  __syncthreads();

  // phase 4: o = gamma * v @ beta + x ; store att[n][f*30+j]
  {
    const int ft = wave >> 1, jt = wave & 1;
    const int f = ft * 16 + a_row(lane);
    const int j = jt * 16 + b_col(lane);
    v8f acc = {};
    v16h a = load_a_row(s_v + f * 32, lane);      // K=i contiguous
    v16h bf = load_b_row(s_btT + j * 32, lane);   // betaT row, K=i contiguous
    acc = wmma16(a, bf, acc);
    const float gm = gamma[0];
    const int jD = jt * 16 + cd_col(lane);
#pragma unroll
    for (int r = 0; r < 8; ++r) {
      int fD = ft * 16 + cd_row(lane, r);
      if (jD < S_) {
        float o = gm * acc[r] + (float)s_xt[jD * F_ + fD];
        att[(long)n * FS_ + fD * S_ + jD] = (_Float16)o;
      }
    }
  }
}

// ---------------- FC1: (3712 x 1920) @ (1920 x 128) + bias, ReLU ----------
__global__ void k_fc1(const _Float16* __restrict__ att,  // [3712][1920]
                      const _Float16* __restrict__ w16,  // [128][1920]
                      const float* __restrict__ bias, _Float16* __restrict__ seqH) {
  const int lane = threadIdx.x;
  const int mt = blockIdx.x;   // 232
  const int nt = blockIdx.y;   // 8
  const _Float16* arow = att + (long)(mt * 16 + a_row(lane)) * FS_;
  const _Float16* brow = w16 + (long)(nt * 16 + b_col(lane)) * FS_;
  v8f acc = {};
  for (int kk = 0; kk < FS_ / 32; ++kk) {
    const int kb0 = kk * 32;
    __builtin_prefetch(arow + kb0 + 512, 0, 0);
    v16h a = load_a_row(arow + kb0, lane);
    v16h bf = load_b_row(brow + kb0, lane);
    acc = wmma16(a, bf, acc);
  }
  const int hh = nt * 16 + cd_col(lane);
#pragma unroll
  for (int r = 0; r < 8; ++r) {
    int rr = mt * 16 + cd_row(lane, r);
    float y = fmaxf(acc[r] + bias[hh], 0.f);
    seqH[(long)rr * H_ + hh] = (_Float16)y;
  }
}

// ---------------- one GRU step (one layer, one timestep) ------------------
// gi = x W_ih^T + b_ih ; gh = h W_hh^T + b_hh ; gates fused in epilogue
__global__ void k_gru_step(const _Float16* __restrict__ xin, int xstride,
                           const _Float16* __restrict__ hprev,      // [128][128]
                           const _Float16* __restrict__ wih16,      // [384][128]
                           const _Float16* __restrict__ whh16,      // [384][128]
                           const float* __restrict__ bih, const float* __restrict__ bhh,
                           _Float16* __restrict__ hnew, _Float16* __restrict__ ysout) {
  const int lane = threadIdx.x;
  const int bt = blockIdx.x;   // 8 batch tiles
  const int ht = blockIdx.y;   // 8 hidden tiles
  const int brow = bt * 16 + a_row(lane);
  const int hcol = ht * 16 + b_col(lane);
  const _Float16* xrow = xin + (long)brow * xstride;
  const _Float16* hrow = hprev + brow * H_;
  v8f gi[3] = {{}, {}, {}}, gh[3] = {{}, {}, {}};
#pragma unroll
  for (int kk = 0; kk < 4; ++kk) {
    const int kb0 = kk * 32;
    v16h ax = load_a_row(xrow + kb0, lane);
    v16h ah = load_a_row(hrow + kb0, lane);
#pragma unroll
    for (int p = 0; p < 3; ++p) {
      const long colbase = (long)(p * H_ + hcol) * H_ + kb0;
      v16h bi = load_b_row(wih16 + colbase, lane);
      v16h bh = load_b_row(whh16 + colbase, lane);
      gi[p] = wmma16(ax, bi, gi[p]);
      gh[p] = wmma16(ah, bh, gh[p]);
    }
  }
  const int h = ht * 16 + cd_col(lane);
#pragma unroll
  for (int r = 0; r < 8; ++r) {
    int b = bt * 16 + cd_row(lane, r);
    float ir = gi[0][r] + bih[h];
    float iz = gi[1][r] + bih[H_ + h];
    float in_ = gi[2][r] + bih[2 * H_ + h];
    float hr = gh[0][r] + bhh[h];
    float hz = gh[1][r] + bhh[H_ + h];
    float hn = gh[2][r] + bhh[2 * H_ + h];
    float rr = 1.f / (1.f + __expf(-(ir + hr)));
    float zz = 1.f / (1.f + __expf(-(iz + hz)));
    float nn = tanhf(in_ + rr * hn);
    float hp = (float)hprev[b * H_ + h];
    float hv = (1.f - zz) * nn + zz * hp;
    hnew[b * H_ + h] = (_Float16)hv;
    ysout[b * H_ + h] = (_Float16)hv;
  }
}

// ---------------- classifier head ----------------
__global__ void k_pred(const _Float16* __restrict__ hlast, const float* __restrict__ pw,
                       const float* __restrict__ pb, float* __restrict__ out) {
  int idx = blockIdx.x * blockDim.x + threadIdx.x;
  if (idx >= B_ * NC_) return;
  int b = idx / NC_, c = idx - (idx / NC_) * NC_;
  float acc = pb[c];
#pragma unroll 8
  for (int h = 0; h < H_; ++h) acc += (float)hlast[b * H_ + h] * pw[c * H_ + h];
  out[idx] = acc;
}

// ---------------- host side ----------------
extern "C" void kernel_launch(void* const* d_in, const int* in_sizes, int n_in,
                              void* d_out, int out_size, void* d_ws, size_t ws_size,
                              hipStream_t stream) {
  const float* x       = (const float*)d_in[0];
  const float* conv0_w = (const float*)d_in[1];
  const float* convs_w = (const float*)d_in[2];
  const float* conv_b  = (const float*)d_in[3];
  const float* bn_g    = (const float*)d_in[4];
  const float* bn_b    = (const float*)d_in[5];
  const float* bn_m    = (const float*)d_in[6];
  const float* bn_v    = (const float*)d_in[7];
  const float* wq      = (const float*)d_in[8];
  const float* wk      = (const float*)d_in[9];
  const float* wv      = (const float*)d_in[10];
  const float* sa_g    = (const float*)d_in[11];
  const float* fc1_w   = (const float*)d_in[12];
  const float* fc1_b   = (const float*)d_in[13];
  const float* g_wih   = (const float*)d_in[14];
  const float* g_whh   = (const float*)d_in[15];
  const float* g_bih   = (const float*)d_in[16];
  const float* g_bhh   = (const float*)d_in[17];
  const float* pred_w  = (const float*)d_in[18];
  const float* pred_b  = (const float*)d_in[19];
  float* out = (float*)d_out;

  // workspace carve-up (f16 elements, 256B aligned)
  char* ws = (char*)d_ws;
  size_t off = 0;
  auto take = [&](size_t elems) {
    char* p = ws + off;
    off += (elems * sizeof(_Float16) + 255) & ~(size_t)255;
    return (_Float16*)p;
  };
  const size_t ACT_MAX = (size_t)F_ * B_ * L0_ * S_;   // biggest activation
  _Float16* actA   = take(ACT_MAX);
  _Float16* actB   = take(ACT_MAX);
  _Float16* wc16   = take((size_t)3 * F_ * KS_ * F_);  // conv1..3 weights
  _Float16* wq16   = take(F_ * F_);
  _Float16* wk16   = take(F_ * F_);
  _Float16* wv16   = take(F_ * F_);
  _Float16* fcw16  = take((size_t)H_ * FS_);
  _Float16* wih16  = take((size_t)2 * 3 * H_ * H_);
  _Float16* whh16  = take((size_t)2 * 3 * H_ * H_);
  _Float16* seqH   = take((size_t)NSL_ * H_);
  _Float16* ys0    = take((size_t)L3_ * B_ * H_);
  _Float16* ys1    = take((size_t)L3_ * B_ * H_);
  _Float16* hbuf0  = take((size_t)B_ * H_);
  _Float16* hbuf1  = take((size_t)B_ * H_);
  _Float16* att    = actA;   // reuse: attention reads actB, writes into actA

  auto cvt = [&](const float* s, _Float16* d, int n) {
    k_cvt_f16<<<(n + 255) / 256, 256, 0, stream>>>(s, d, n);
  };

  // weight prep (f32 -> f16)
  cvt(wq, wq16, F_ * F_);
  cvt(wk, wk16, F_ * F_);
  cvt(wv, wv16, F_ * F_);
  cvt(fc1_w, fcw16, H_ * FS_);
  cvt(g_wih, wih16, 2 * 3 * H_ * H_);
  cvt(g_whh, whh16, 2 * 3 * H_ * H_);
  for (int l = 0; l < 3; ++l) {
    k_convw<<<(F_ * KS_ * F_ + 255) / 256, 256, 0, stream>>>(
        convs_w + (size_t)l * F_ * F_ * KS_, wc16 + (size_t)l * F_ * KS_ * F_);
  }

  // conv0
  {
    int n = B_ * L0_ * S_;
    k_conv0<<<(n + 255) / 256, 256, 0, stream>>>(
        x, conv0_w, conv_b, bn_g, bn_b, bn_m, bn_v, actA);
  }
  // conv1..3 as WMMA implicit GEMMs (ping-pong actA/actB)
  {
    dim3 g1((B_ * L1_ * S_) / 16, 4);
    k_conv_wmma<L0_, L1_><<<g1, 32, 0, stream>>>(
        actA, wc16 + 0 * F_ * KS_ * F_,
        conv_b + 1 * F_, bn_g + 1 * F_, bn_b + 1 * F_, bn_m + 1 * F_, bn_v + 1 * F_,
        actB);
    dim3 g2((B_ * L2_ * S_) / 16, 4);
    k_conv_wmma<L1_, L2_><<<g2, 32, 0, stream>>>(
        actB, wc16 + 1 * F_ * KS_ * F_,
        conv_b + 2 * F_, bn_g + 2 * F_, bn_b + 2 * F_, bn_m + 2 * F_, bn_v + 2 * F_,
        actA);
    dim3 g3((B_ * L3_ * S_) / 16, 4);
    k_conv_wmma<L2_, L3_><<<g3, 32, 0, stream>>>(
        actA, wc16 + 2 * F_ * KS_ * F_,
        conv_b + 3 * F_, bn_g + 3 * F_, bn_b + 3 * F_, bn_m + 3 * F_, bn_v + 3 * F_,
        actB);
  }
  // fused self-attention: conv3 output is in actB; write att into actA
  k_attn<<<NSL_, 256, 0, stream>>>(actB, wq16, wk16, wv16, sa_g, att);

  // FC1
  {
    dim3 grid(NSL_ / 16, H_ / 16);
    k_fc1<<<grid, 32, 0, stream>>>(att, fcw16, fc1_b, seqH);
  }

  // 2-layer GRU, 29 steps each; h ping-pongs in f16
  {
    _Float16* hb[2] = {hbuf0, hbuf1};
    dim3 grid(B_ / 16, H_ / 16);
    // layer 0: input rows are seqH[(b*L + t)*H], stride L*H
    k_zero_f16<<<(B_ * H_ + 255) / 256, 256, 0, stream>>>(hbuf0, B_ * H_);
    for (int t = 0; t < L3_; ++t) {
      k_gru_step<<<grid, 32, 0, stream>>>(
          seqH + (size_t)t * H_, L3_ * H_, hb[t & 1],
          wih16, whh16, g_bih, g_bhh,
          hb[(t + 1) & 1], ys0 + (size_t)t * B_ * H_);
    }
    // layer 1: input is ys0[t] contiguous [B][H]
    k_zero_f16<<<(B_ * H_ + 255) / 256, 256, 0, stream>>>(hbuf0, B_ * H_);
    for (int t = 0; t < L3_; ++t) {
      k_gru_step<<<grid, 32, 0, stream>>>(
          ys0 + (size_t)t * B_ * H_, H_, hb[t & 1],
          wih16 + 3 * H_ * H_, whh16 + 3 * H_ * H_, g_bih + 3 * H_, g_bhh + 3 * H_,
          hb[(t + 1) & 1], ys1 + (size_t)t * B_ * H_);
    }
  }

  // head: last timestep of layer-1 output
  k_pred<<<(B_ * NC_ + 255) / 256, 256, 0, stream>>>(
      ys1 + (size_t)(L3_ - 1) * B_ * H_, pred_w, pred_b, out);
}
